// LinearAttention_56676388438112
// MI455X (gfx1250) — matrix operand order repaired and checked by reading, not compiled, not run on hardware
//
#include <hip/hip_runtime.h>
#include <hip/hip_bf16.h>

// Problem dims (fixed by reference)
#define SDIM 4096
#define BDIM 8
#define EDIM 512
#define HDIM 8
#define DDIM 64
#define NROW (SDIM * BDIM)   // 32768 rows of the (S*B, E) activation matrices

typedef __bf16 bf16_t;
typedef __attribute__((ext_vector_type(16))) __bf16 v16bf;
typedef __attribute__((ext_vector_type(2)))  __bf16 v2bf;
typedef __attribute__((ext_vector_type(8)))  float  v8f;
typedef __attribute__((ext_vector_type(4)))  unsigned int u32x4;
typedef __attribute__((ext_vector_type(8)))  int i32x8;
typedef __attribute__((ext_vector_type(4)))  int i32x4;

#if __has_builtin(__builtin_amdgcn_tensor_load_to_lds) && __has_builtin(__builtin_amdgcn_s_wait_tensorcnt)
#define HAS_TDM 1
#endif

// ---------------------------------------------------------------------------
// packed f32 -> bf16 (v_cvt_pk_bf16_f32 when available)
// ---------------------------------------------------------------------------
__device__ inline v2bf pk_bf16(float a, float b) {
#if __has_builtin(__builtin_amdgcn_cvt_pk_bf16_f32)
    return __builtin_bit_cast(v2bf, __builtin_amdgcn_cvt_pk_bf16_f32(a, b));
#else
    v2bf r; r[0] = (bf16_t)a; r[1] = (bf16_t)b; return r;
#endif
}

// ---------------------------------------------------------------------------
// WMMA bf16 16x16x32 fragment helpers (wave32 layouts per CDNA5 ISA 7.12.2)
//   A (16x32 MxK): lane<16 -> row M=lane,    elems = K{0..7},K{16..23}
//                  lane>=16 -> row M=lane-16, elems = K{8..15},K{24..31}
//   B (32x16 KxN): lane<16 -> col N=lane, elems K{0..7},K{16..23} ; lane>=16 -> K{8..15},K{24..31}
//   C/D (16x16):   lane col N=lane%16, rows M = (lane>=16?8:0)+vgpr
// ---------------------------------------------------------------------------
__device__ inline v16bf load_a_f32(const float* __restrict__ row, int k0, int hk) {
    const float4* p0 = reinterpret_cast<const float4*>(row + k0 + hk * 8);
    const float4* p1 = reinterpret_cast<const float4*>(row + k0 + 16 + hk * 8);
    float4 x0 = p0[0], x1 = p0[1], y0 = p1[0], y1 = p1[1];
    union { v2bf p[8]; v16bf v; } t;
    t.p[0] = pk_bf16(x0.x, x0.y); t.p[1] = pk_bf16(x0.z, x0.w);
    t.p[2] = pk_bf16(x1.x, x1.y); t.p[3] = pk_bf16(x1.z, x1.w);
    t.p[4] = pk_bf16(y0.x, y0.y); t.p[5] = pk_bf16(y0.z, y0.w);
    t.p[6] = pk_bf16(y1.x, y1.y); t.p[7] = pk_bf16(y1.z, y1.w);
    return t.v;
}

__device__ inline v16bf load_a_bf16(const bf16_t* __restrict__ row, int k0, int hk) {
    union { uint4 u[2]; v16bf v; } t;
    t.u[0] = *reinterpret_cast<const uint4*>(row + k0 + hk * 8);
    t.u[1] = *reinterpret_cast<const uint4*>(row + k0 + 16 + hk * 8);
    return t.v;
}

__device__ inline v16bf load_b_bf16(const bf16_t* __restrict__ colrow, int k0, int hk) {
    union { uint4 u[2]; v16bf v; } t;
    t.u[0] = *reinterpret_cast<const uint4*>(colrow + k0 + hk * 8);
    t.u[1] = *reinterpret_cast<const uint4*>(colrow + k0 + 16 + hk * 8);
    return t.v;
}

// ---------------------------------------------------------------------------
// K0: fp32 -> bf16 weight conversion
// ---------------------------------------------------------------------------
__global__ void cvt_f32_bf16(const float* __restrict__ src, bf16_t* __restrict__ dst, int n) {
    int idx = (blockIdx.x * blockDim.x + threadIdx.x) * 4;
    if (idx < n) {
        float4 v = *reinterpret_cast<const float4*>(src + idx);
        union { uint2 u; v2bf p[2]; } pk;
        pk.p[0] = pk_bf16(v.x, v.y);
        pk.p[1] = pk_bf16(v.z, v.w);
        *reinterpret_cast<uint2*>(dst + idx) = pk.u;
    }
}

// ---------------------------------------------------------------------------
// K1/K4: Y = epilogue( X @ W^T )  with X:(NROW,512) W:(512,512) row-major (n,k)
// MODE 0: q = (elu(y)+1)*D^-0.5 -> bf16      MODE 1: k = mask(elu(y)+1) -> bf16
// MODE 2: v = y -> bf16                      MODE 3: out = y -> f32 (X is bf16 ctx)
// Block 256 thr = 8 waves; block tile 128x128; wave tile 64x32 (4x2 WMMA tiles)
// ---------------------------------------------------------------------------
template <int MODE>
__global__ __launch_bounds__(256) void proj_gemm(
    const float* __restrict__ X, const bf16_t* __restrict__ Xb,
    const bf16_t* __restrict__ W, bf16_t* __restrict__ Yb,
    float* __restrict__ Yf, const unsigned char* __restrict__ mask) {
    const int tid  = threadIdx.x;
    const int lane = tid & 31, lr = lane & 15, hk = lane >> 4;
    const int w = tid >> 5, wm = w >> 2, wn = w & 3;
    const int rowbase = blockIdx.x * 128 + wm * 64;
    const int colbase = blockIdx.y * 128 + wn * 32;

    v8f acc[4][2] = {};

    for (int k0 = 0; k0 < EDIM; k0 += 32) {
        v16bf a[4], bm[2];
#pragma unroll
        for (int ms = 0; ms < 4; ++ms) {
            const int row = rowbase + ms * 16 + lr;
            if (MODE == 3) {
                a[ms] = load_a_bf16(Xb + (long)row * EDIM, k0, hk);
            } else {
                const float* xr = X + (long)row * EDIM;
                if (k0 + 32 < EDIM) __builtin_prefetch(xr + k0 + 32, 0, 0);
                a[ms] = load_a_f32(xr, k0, hk);
            }
        }
#pragma unroll
        for (int ns = 0; ns < 2; ++ns) {
            const int col = colbase + ns * 16 + lr;
            bm[ns] = load_b_bf16(W + (long)col * EDIM, k0, hk);
        }
#pragma unroll
        for (int ms = 0; ms < 4; ++ms)
#pragma unroll
            for (int ns = 0; ns < 2; ++ns)
                acc[ms][ns] = __builtin_amdgcn_wmma_f32_16x16x32_bf16(
                    false, a[ms], false, bm[ns], (short)0, acc[ms][ns], false, false);
    }

#pragma unroll
    for (int ms = 0; ms < 4; ++ms)
#pragma unroll
        for (int ns = 0; ns < 2; ++ns) {
            const int col = colbase + ns * 16 + lr;
#pragma unroll
            for (int j = 0; j < 8; ++j) {
                const int row = rowbase + ms * 16 + hk * 8 + j;
                float v = acc[ms][ns][j];
                if (MODE == 0) v = (v > 0.f) ? (v + 1.f) * 0.125f : __expf(v) * 0.125f;
                if (MODE == 1) {
                    v = (v > 0.f) ? (v + 1.f) : __expf(v);
                    const int b = row & (BDIM - 1), s = row >> 3;  // row = s*B + b
                    if (mask[(long)b * SDIM + s]) v = 0.f;        // numpy bool = 1 byte
                }
                if (MODE == 3)
                    Yf[(long)row * EDIM + col] = v;
                else
                    Yb[(long)row * EDIM + col] = (bf16_t)v;
            }
        }
}

// ---------------------------------------------------------------------------
// K2: per (b,h)  kv = K^T @ V  (64x64, reduce over S=4096) + k_sum
// K/V tiles are transposed at staging time into LDS as [d][s] (row stride 40
// bf16 = 80B, multiple of 16B), so every WMMA fragment read is a pair of
// vectorized ds_load_b128 instead of 16 scalar ds_load_u16.
// Stores kv transposed as kvT[bh][vd][kd] so K3's B-fragments are contiguous.
// ---------------------------------------------------------------------------
__global__ __launch_bounds__(256) void kv_kernel(
    const bf16_t* __restrict__ kb, const bf16_t* __restrict__ vb,
    bf16_t* __restrict__ kvT, float* __restrict__ ksum) {
    __shared__ bf16_t lkT[64][40];   // [d][s], 8-elem pad keeps rows 16B aligned
    __shared__ bf16_t lvT[64][40];
    const int tid = threadIdx.x, lane = tid & 31, lr = lane & 15, hk = lane >> 4, w = tid >> 5;
    const int bh = blockIdx.x, b = bh / HDIM, h = bh % HDIM;
    const int sl = tid >> 3, d0 = (tid & 7) * 8;     // 256 thr x uint4 = 32x64 tile
    const long base = (long)b * EDIM + (long)h * DDIM;

    v8f acc[2] = {};
    float ks = 0.f;

    for (int s0 = 0; s0 < SDIM; s0 += 32) {
        const long goff = (long)(s0 + sl) * BDIM * EDIM + base + d0;
        union { uint4 u; bf16_t h[8]; } tk, tv;
        tk.u = *reinterpret_cast<const uint4*>(kb + goff);
        tv.u = *reinterpret_cast<const uint4*>(vb + goff);
#pragma unroll
        for (int i = 0; i < 8; ++i) {                // in-register transpose -> LDS
            lkT[d0 + i][sl] = tk.h[i];
            lvT[d0 + i][sl] = tv.h[i];
        }
        __syncthreads();

        if (tid < DDIM) {   // k_sum: contiguous row of lkT, runs alongside WMMA work
#pragma unroll
            for (int s = 0; s < 32; ++s) ks += (float)lkT[tid][s];
        }
#pragma unroll
        for (int t = 0; t < 2; ++t) {   // 16 output tiles / 8 waves
            const int ti = w * 2 + t, mi = ti >> 2, ni = ti & 3;
            const int kd = mi * 16 + lr, vd = ni * 16 + lr;
            union { uint4 u[2]; v16bf v; } a, bfr;
            a.u[0]   = *reinterpret_cast<const uint4*>(&lkT[kd][hk * 8]);        // s 0..7 / 8..15
            a.u[1]   = *reinterpret_cast<const uint4*>(&lkT[kd][16 + hk * 8]);   // s 16..23 / 24..31
            bfr.u[0] = *reinterpret_cast<const uint4*>(&lvT[vd][hk * 8]);
            bfr.u[1] = *reinterpret_cast<const uint4*>(&lvT[vd][16 + hk * 8]);
            acc[t] = __builtin_amdgcn_wmma_f32_16x16x32_bf16(
                false, a.v, false, bfr.v, (short)0, acc[t], false, false);
        }
        __syncthreads();
    }

#pragma unroll
    for (int t = 0; t < 2; ++t) {
        const int ti = w * 2 + t, mi = ti >> 2, ni = ti & 3;
        const int kd0 = mi * 16 + hk * 8, vd = ni * 16 + lr;
        union { uint4 u; v2bf p[4]; } pk;
#pragma unroll
        for (int j = 0; j < 4; ++j) pk.p[j] = pk_bf16(acc[t][2 * j], acc[t][2 * j + 1]);
        *reinterpret_cast<uint4*>(kvT + (long)bh * DDIM * DDIM + (long)vd * DDIM + kd0) = pk.u;
    }
    if (tid < DDIM) ksum[bh * DDIM + tid] = ks;
}

// ---------------------------------------------------------------------------
// K3: per (b,h): ctx = (Q @ kv) / (Q . k_sum + 1e-5). Block tile 128 s-rows x 64 vd.
// kv (8KB bf16) staged into LDS by the Tensor Data Mover when available
// (D#: 1-D 8192B tile, data_size=2B, pad_enable inserts 16B per 128B row so the
// hardware reproduces the padded [64][72] layout); manual copy otherwise.
// ---------------------------------------------------------------------------
__global__ __launch_bounds__(256) void ctx_kernel(
    const bf16_t* __restrict__ qb, const bf16_t* __restrict__ kvT,
    const float* __restrict__ ksum, bf16_t* __restrict__ ctxb) {
    __shared__ bf16_t lkv[64][72];   // row stride 144B (16B multiple)
    __shared__ float nrm_s[128];
    const int tid = threadIdx.x, lane = tid & 31, lr = lane & 15, hk = lane >> 4, w = tid >> 5;
    const int wm = w >> 1, wn = w & 1;   // 4 M-waves x 2 N-waves
    const int s0 = blockIdx.x * 128, b = blockIdx.y, h = blockIdx.z;
    const int bh = b * HDIM + h;
    const long qbase = (long)b * EDIM + (long)h * DDIM;

#ifdef HAS_TDM
    if (tid < 32) {   // one wave drives the TDM (EXEC ignored by tensor ops)
        const unsigned lds_addr = (unsigned)(unsigned long long)(&lkv[0][0]);
        const unsigned long long ga =
            (unsigned long long)(kvT + (long)bh * DDIM * DDIM);
        u32x4 g0;
        g0[0] = 1u;                                            // count=1 (valid D#)
        g0[1] = lds_addr;                                      // lds_addr
        g0[2] = (unsigned)(ga & 0xFFFFFFFFull);                // global_addr[31:0]
        g0[3] = (unsigned)((ga >> 32) & 0x1FFFFFFull) | 0x80000000u;  // ga[56:32] | type=2
        i32x8 g1;
        // data_size=2B | pad_enable | pad_interval=32dw(128B) | pad_amount=4dw(16B)
        g1[0] = (int)0x07110000;
        g1[1] = (int)((4096u & 0xFFFFu) << 16);  // tensor_dim0 = 4096 elems (lo16)
        g1[2] = (int)(1u << 16);                 // tensor_dim0 hi=0 | tensor_dim1 = 1
        g1[3] = (int)(4096u << 16);              // tile_dim0 = 4096 elems (1-D tile)
        g1[4] = 0;                               // tile_dim1=0, tile_dim2=0
        g1[5] = 4096;                            // tensor_dim0_stride (lo32)
        g1[6] = 0;
        g1[7] = 0;
        i32x4 z4 = {0, 0, 0, 0};
        i32x8 z8 = {0, 0, 0, 0, 0, 0, 0, 0};
        // clang-23 / amdgpu-toolchain form: 6 args (g0, g1, g2, g3, g4, cpol)
        __builtin_amdgcn_tensor_load_to_lds(g0, g1, z4, z4, z8, 0);
        __builtin_amdgcn_s_wait_tensorcnt(0);
    }
#else
#pragma unroll
    for (int j = 0; j < 2; ++j) {   // manual: 512 uint4 / 256 thr
        const int e = tid * 2 + j, vd = e >> 3, kd = (e & 7) * 8;
        *reinterpret_cast<uint4*>(&lkv[vd][kd]) =
            *reinterpret_cast<const uint4*>(kvT + (long)bh * DDIM * DDIM + (long)vd * DDIM + kd);
    }
#endif
    if (tid < 128) {                // per-row norm = q . k_sum
        const bf16_t* qr = qb + (long)(s0 + tid) * BDIM * EDIM + qbase;
        const float* kr = ksum + bh * DDIM;
        float nv = 0.f;
#pragma unroll
        for (int kd = 0; kd < DDIM; ++kd) nv += (float)qr[kd] * kr[kd];
        nrm_s[tid] = nv + 1e-5f;
    }
    __syncthreads();

    v8f acc[2][2] = {};
#pragma unroll
    for (int kk = 0; kk < 2; ++kk) {
        const int k0 = kk * 32;
        v16bf a[2], bm[2];
#pragma unroll
        for (int ms = 0; ms < 2; ++ms) {
            const int rt = wm * 32 + ms * 16 + lr;
            a[ms] = load_a_bf16(qb + (long)(s0 + rt) * BDIM * EDIM + qbase, k0, hk);
        }
#pragma unroll
        for (int ns = 0; ns < 2; ++ns) {
            const int vd = wn * 32 + ns * 16 + lr;
            union { uint4 u[2]; v16bf v; } t;
            t.u[0] = *reinterpret_cast<const uint4*>(&lkv[vd][k0 + hk * 8]);
            t.u[1] = *reinterpret_cast<const uint4*>(&lkv[vd][k0 + 16 + hk * 8]);
            bm[ns] = t.v;
        }
#pragma unroll
        for (int ms = 0; ms < 2; ++ms)
#pragma unroll
            for (int ns = 0; ns < 2; ++ns)
                acc[ms][ns] = __builtin_amdgcn_wmma_f32_16x16x32_bf16(
                    false, a[ms], false, bm[ns], (short)0, acc[ms][ns], false, false);
    }

#pragma unroll
    for (int ms = 0; ms < 2; ++ms)
#pragma unroll
        for (int ns = 0; ns < 2; ++ns) {
            const int vd = wn * 32 + ns * 16 + lr;
#pragma unroll
            for (int j = 0; j < 8; ++j) {
                const int rt = wm * 32 + ms * 16 + hk * 8 + j;
                const float v = acc[ms][ns][j] / nrm_s[rt];
                ctxb[((long)(s0 + rt) * BDIM + b) * EDIM + (long)h * DDIM + vd] = (bf16_t)v;
            }
        }
}

// ---------------------------------------------------------------------------
// Host-side orchestration
// ---------------------------------------------------------------------------
extern "C" void kernel_launch(void* const* d_in, const int* in_sizes, int n_in,
                              void* d_out, int out_size, void* d_ws, size_t ws_size,
                              hipStream_t stream) {
    const float* query = (const float*)d_in[0];
    const float* key_t = (const float*)d_in[1];
    const float* value = (const float*)d_in[2];
    const unsigned char* mask = (const unsigned char*)d_in[3];  // bool[B][S]
    const float* Wq = (const float*)d_in[4];
    const float* Wk = (const float*)d_in[5];
    const float* Wv = (const float*)d_in[6];
    const float* Wp = (const float*)d_in[7];
    float* out = (float*)d_out;

    const long EE  = (long)EDIM * EDIM;         // 262144
    const long NE  = (long)NROW * EDIM;         // 16.7M elements
    char* ws = (char*)d_ws;
    bf16_t* wb   = (bf16_t*)ws;                                  // [4][E*E] bf16
    bf16_t* qb   = (bf16_t*)(ws + 4 * EE * 2);                   // [N][E] bf16
    bf16_t* kb   = (bf16_t*)(ws + 4 * EE * 2 + NE * 2);
    bf16_t* vb   = (bf16_t*)(ws + 4 * EE * 2 + NE * 4);
    bf16_t* ctxb = (bf16_t*)(ws + 4 * EE * 2 + NE * 6);
    bf16_t* kvT  = (bf16_t*)(ws + 4 * EE * 2 + NE * 8);          // [64][64][64] bf16
    float*  ksum = (float*) (ws + 4 * EE * 2 + NE * 8 + (long)BDIM * HDIM * DDIM * DDIM * 2);

    // K0: weights -> bf16
    const int cvtBlocks = (int)(EE / 4 / 256);
    cvt_f32_bf16<<<cvtBlocks, 256, 0, stream>>>(Wq, wb + 0 * EE, (int)EE);
    cvt_f32_bf16<<<cvtBlocks, 256, 0, stream>>>(Wk, wb + 1 * EE, (int)EE);
    cvt_f32_bf16<<<cvtBlocks, 256, 0, stream>>>(Wv, wb + 2 * EE, (int)EE);
    cvt_f32_bf16<<<cvtBlocks, 256, 0, stream>>>(Wp, wb + 3 * EE, (int)EE);

    // K1: projections with fused activations/mask
    dim3 gGemm(NROW / 128, EDIM / 128);
    proj_gemm<0><<<gGemm, 256, 0, stream>>>(query, nullptr, wb + 0 * EE, qb, nullptr, nullptr);
    proj_gemm<1><<<gGemm, 256, 0, stream>>>(key_t, nullptr, wb + 1 * EE, kb, nullptr, mask);
    proj_gemm<2><<<gGemm, 256, 0, stream>>>(value, nullptr, wb + 2 * EE, vb, nullptr, nullptr);

    // K2: kv state + k_sum per (b,h)
    kv_kernel<<<BDIM * HDIM, 256, 0, stream>>>(kb, vb, kvT, ksum);

    // K3: ctx = (Q @ kv) / (Q . k_sum + eps)
    ctx_kernel<<<dim3(SDIM / 128, BDIM, HDIM), 256, 0, stream>>>(qb, kvT, ksum, ctxb);

    // K4: output projection -> f32 d_out
    proj_gemm<3><<<gGemm, 256, 0, stream>>>(nullptr, ctxb, wb + 3 * EE, nullptr, out, nullptr);
}
